// MAMLTraineeOmniglotConv_12575664243447
// MI455X (gfx1250) — compile-verified, matrix operand order checked by reading, not run on hardware
//
#include <hip/hip_runtime.h>

// ---------------------------------------------------------------------------
// MAML Omniglot conv stack for MI455X (gfx1250, wave32, WMMA).
// Layers 2-4: implicit-GEMM conv with v_wmma_f32_16x16x32_f16 (K=576).
// Readout:    per-task GEMM with v_wmma_f32_16x16x4_f32.
// BatchNorm:  global per-channel stats via LDS + global float atomics.
// ---------------------------------------------------------------------------

typedef _Float16 v16h __attribute__((ext_vector_type(16)));
typedef float    v8f  __attribute__((ext_vector_type(8)));
typedef float    v2f  __attribute__((ext_vector_type(2)));
typedef unsigned v8u  __attribute__((ext_vector_type(8)));

#define BN_EPS 1e-3f

// ------------------------------------------------------------------ stats ---
__global__ void zero_stats_kernel(float* __restrict__ stats) {
    int i = blockIdx.x * 256 + threadIdx.x;
    if (i < 512) stats[i] = 0.f;
}

// ------------------------------------------------------- layer 1 (Cin = 1) --
// x: [64,64,28,28,1] f32, k1: [64,3,3,1,64] f32 -> h1 raw f16 [64,64,14,14,64]
__global__ void conv1_kernel(const float* __restrict__ x,
                             const float* __restrict__ k1,
                             _Float16* __restrict__ h1,
                             float* __restrict__ stats) {
    __shared__ float s_sum[128];
    const int tid = threadIdx.x;
    if (tid < 128) s_sum[tid] = 0.f;
    __syncthreads();

    long idx = (long)blockIdx.x * 256 + tid;      // [b][e][oy][ox][co]
    const int co = (int)(idx & 63);
    long r = idx >> 6;
    const int ox = (int)(r % 14); r /= 14;
    const int oy = (int)(r % 14); r /= 14;
    const int e  = (int)(r & 63);
    const int b  = (int)(r >> 6);

    const float* xb = x  + (((long)b * 64 + e) * 28) * 28;   // [28][28]
    const float* kb = k1 + (long)b * 9 * 64;                 // [ky*3+kx][co]

    float y = 0.f;
#pragma unroll
    for (int ky = 0; ky < 3; ++ky) {
        const int iy = 2 * oy - 1 + ky;
        if ((unsigned)iy >= 28u) continue;
#pragma unroll
        for (int kx = 0; kx < 3; ++kx) {
            const int ix = 2 * ox - 1 + kx;
            if ((unsigned)ix >= 28u) continue;
            y += xb[iy * 28 + ix] * kb[(ky * 3 + kx) * 64 + co];
        }
    }
    y = fmaxf(y, 0.f);
    h1[idx] = (_Float16)y;

    atomicAdd(&s_sum[co], y);
    atomicAdd(&s_sum[64 + co], y * y);
    __syncthreads();
    if (tid < 128) atomicAdd(&stats[tid], s_sum[tid]);
}

// --------------------------------------------------------- BN normalize -----
// In-place (y - mean) * rsqrt(var + eps) on an f16 buffer, channels innermost.
__global__ void norm_kernel(_Float16* __restrict__ h,
                            const float* __restrict__ stats,
                            float inv_count, long n) {
    long idx = (long)blockIdx.x * 256 + threadIdx.x;
    if (idx >= n) return;
    const int c = (int)(idx & 63);
    const float m   = stats[c] * inv_count;
    const float var = stats[64 + c] * inv_count - m * m;
    const float s   = rsqrtf(var + BN_EPS);
    h[idx] = (_Float16)(((float)h[idx] - m) * s);
}

// -------------------------------------------- layers 2-4: WMMA conv GEMM ----
// Per task: M = 64*S_OUT^2 rows, N = 64 channels, K = 576 (3*3*64).
// Block = 256 threads = 8 waves; block tile = 32 M x 64 N.
// Wave (m_sub = w>>2, n_sub = w&3) owns one 16x16 D tile, K-loop of 18 WMMAs.
template <int S_IN, int S_OUT>
__global__ void convw_kernel(const _Float16* __restrict__ hin,
                             const float* __restrict__ w,
                             _Float16* __restrict__ hout,
                             float* __restrict__ stats) {
    __shared__ _Float16 wt[64 * 64];   // weight chunk, [n][k_local] f16 (8 KB)
    __shared__ float    s_sum[128];

    const int tid = threadIdx.x;
    const int b   = blockIdx.y;
    if (tid < 128) s_sum[tid] = 0.f;

    const int wave = tid >> 5, lane = tid & 31;
    const int half = lane >> 4, l16 = lane & 15;
    const int m_sub = wave >> 2, n_sub = wave & 3;
    const int m_frag = blockIdx.x * 32 + m_sub * 16;

    // Decode this lane's A row (row = lane%16 of the fragment).
    const int SS   = S_OUT * S_OUT;
    const int mrow = m_frag + l16;
    const int e    = mrow / SS;
    const int rem  = mrow - e * SS;
    const int oy   = rem / S_OUT;
    const int ox   = rem - oy * S_OUT;
    const int iy0  = 2 * oy - 1, ix0 = 2 * ox - 1;

    const _Float16* inb = hin + (long)b * 64 * S_IN * S_IN * 64; // [e][y][x][c]
    const float*    wb  = w   + (long)b * 576 * 64;              // [k][n]
    const int n_col = n_sub * 16 + l16;
    const _Float16* wl = &wt[n_col * 64];

    v8f acc = {};
    for (int kc = 0; kc < 576; kc += 64) {           // 9 weight chunks
        __syncthreads();
        // Coalesced global read w[kc*64 + i]; LDS transposed to [n][k_local].
        for (int i = tid; i < 4096; i += 256)
            wt[((i & 63) << 6) | (i >> 6)] = (_Float16)wb[kc * 64 + i];
        __syncthreads();

#pragma unroll
        for (int s = 0; s < 64; s += 32) {           // 2 WMMA steps / chunk
            v8u au, bu;
#pragma unroll
            for (int j = 0; j < 8; ++j) {
                // ISA 16-bit A/B fragment K pattern (pairs along K):
                const int kl = s + 2 * j + ((j >= 4) ? 8 : 0) + 8 * half;
                bu[j] = *(const unsigned*)(wl + kl);          // ds_load_b32

                const int k   = kc + kl;
                const int tap = k >> 6, ci = k & 63;
                const int ty  = tap / 3;
                const int iy  = iy0 + ty, ix = ix0 + (tap - ty * 3);
                unsigned ap = 0u;
                if ((unsigned)iy < (unsigned)S_IN && (unsigned)ix < (unsigned)S_IN)
                    ap = *(const unsigned*)(inb +
                            (((long)e * S_IN + iy) * S_IN + ix) * 64 + ci);
                au[j] = ap;
            }
            v16h a  = __builtin_bit_cast(v16h, au);
            v16h bm = __builtin_bit_cast(v16h, bu);
            acc = __builtin_amdgcn_wmma_f32_16x16x32_f16(
                    false, a, false, bm, (short)0, acc, false, false);
        }
    }

    // Fused ReLU + store raw f16 + per-channel BN stats.
    float psum = 0.f, psq = 0.f;
#pragma unroll
    for (int rr = 0; rr < 8; ++rr) {
        const float v  = fmaxf(acc[rr], 0.f);
        const int m    = m_frag + rr + 8 * half;     // D: M = r + 8*(lane/16)
        const int e2   = m / SS;
        const int rem2 = m - e2 * SS;
        const int oy2  = rem2 / S_OUT, ox2 = rem2 - oy2 * S_OUT;
        hout[((((long)b * 64 + e2) * S_OUT + oy2) * S_OUT + ox2) * 64 + n_col] =
            (_Float16)v;
        psum += v; psq += v * v;
    }
    atomicAdd(&s_sum[n_col], psum);
    atomicAdd(&s_sum[64 + n_col], psq);
    __syncthreads();
    if (tid < 128) atomicAdd(&stats[tid], s_sum[tid]);
}

// ----------------------------------------- max-pool + BN of layer 4 ---------
// feat[b,e,c] = (max over 2x2 of h4_raw) normalized; BN is monotone so
// normalizing the max equals max of normalized.
__global__ void maxpool_kernel(const _Float16* __restrict__ h4,
                               const float* __restrict__ stats,
                               float* __restrict__ feat) {
    long idx = (long)blockIdx.x * 256 + threadIdx.x;   // B*E*64 = 262144
    const int c = (int)(idx & 63);
    const long be = idx >> 6;
    const _Float16* p = h4 + be * 4 * 64 + c;          // [2][2][64]
    const float v0 = (float)p[0],   v1 = (float)p[64];
    const float v2 = (float)p[128], v3 = (float)p[192];
    const float mx = fmaxf(fmaxf(v0, v1), fmaxf(v2, v3));
    const float inv = 1.f / 16384.f;                   // B*E*2*2
    const float m   = stats[c] * inv;
    const float var = stats[64 + c] * inv - m * m;
    feat[idx] = (mx - m) * rsqrtf(var + BN_EPS);
}

// ------------------------------ per-task readout: f32 WMMA 16x16x4 ----------
// feat[b]: [64,64] f32, w_ro[b]: [64,20] -> out[b]: [64,20].
__global__ void readout_kernel(const float* __restrict__ feat,
                               const float* __restrict__ w_ro,
                               float* __restrict__ out) {
    __shared__ float sf[64 * 64];   // A tile [m][k]
    __shared__ float sw[64 * 32];   // B tile [k][n], N padded 20 -> 32
    const int b = blockIdx.x, tid = threadIdx.x;

    const float* fb = feat + (long)b * 64 * 64;
    for (int i = tid; i < 4096; i += 256) sf[i] = fb[i];
    const float* wb = w_ro + (long)b * 64 * 20;
    for (int i = tid; i < 2048; i += 256) {
        const int k = i >> 5, n = i & 31;
        sw[i] = (n < 20) ? wb[k * 20 + n] : 0.f;
    }
    __syncthreads();

    const int wave = tid >> 5, lane = tid & 31;
    const int half = lane >> 4, l16 = lane & 15;
    const int m_tile = (wave >> 1) * 16, n_tile = (wave & 1) * 16;

    v8f acc = {};
#pragma unroll
    for (int kk = 0; kk < 64; kk += 4) {               // 16 WMMAs, K=4 each
        const int k0 = kk + 2 * half;                  // f32 A: K = 2*(lane/16)+r
        v2f a, bm;
        a[0]  = sf[(m_tile + l16) * 64 + k0];
        a[1]  = sf[(m_tile + l16) * 64 + k0 + 1];
        bm[0] = sw[k0 * 32 + n_tile + l16];
        bm[1] = sw[(k0 + 1) * 32 + n_tile + l16];
        acc = __builtin_amdgcn_wmma_f32_16x16x4_f32(
                false, a, false, bm, (short)0, acc, false, false);
    }

    const int n = n_tile + l16;
    if (n < 20) {
#pragma unroll
        for (int r = 0; r < 8; ++r) {
            const int m = m_tile + r + 8 * half;
            out[((long)b * 64 + m) * 20 + n] = acc[r];
        }
    }
}

// ---------------------------------------------------------------------------
extern "C" void kernel_launch(void* const* d_in, const int* in_sizes, int n_in,
                              void* d_out, int out_size, void* d_ws, size_t ws_size,
                              hipStream_t stream) {
    const float* x    = (const float*)d_in[0];
    const float* k1   = (const float*)d_in[1];
    const float* k2   = (const float*)d_in[2];
    const float* k3   = (const float*)d_in[3];
    const float* k4   = (const float*)d_in[4];
    const float* w_ro = (const float*)d_in[5];
    float* out = (float*)d_out;

    // Workspace carve-up (~134 MB): stats | h1 | h2 | h3 | h4 | feat
    char* ws = (char*)d_ws;
    float* stats = (float*)ws;                         // 4 layers x 128 floats
    const long n1 = 64L * 64 * 14 * 14 * 64;           // 51,380,224
    const long n2 = 64L * 64 * 7 * 7 * 64;             // 12,845,056
    const long n3 = 64L * 64 * 4 * 4 * 64;             //  4,194,304
    const long n4 = 64L * 64 * 2 * 2 * 64;             //  1,048,576
    _Float16* h1 = (_Float16*)(ws + 4096);
    _Float16* h2 = (_Float16*)((char*)h1 + 2 * n1);
    _Float16* h3 = (_Float16*)((char*)h2 + 2 * n2);
    _Float16* h4 = (_Float16*)((char*)h3 + 2 * n3);
    float*  feat = (float*)((char*)h4 + 2 * n4);

    zero_stats_kernel<<<2, 256, 0, stream>>>(stats);

    conv1_kernel<<<(unsigned)(n1 / 256), 256, 0, stream>>>(x, k1, h1, stats + 0 * 128);
    norm_kernel <<<(unsigned)(n1 / 256), 256, 0, stream>>>(h1, stats + 0 * 128,
                                                           1.f / 802816.f, n1);

    convw_kernel<14, 7><<<dim3(98, 64), 256, 0, stream>>>(h1, k2, h2, stats + 1 * 128);
    norm_kernel <<<(unsigned)(n2 / 256), 256, 0, stream>>>(h2, stats + 1 * 128,
                                                           1.f / 200704.f, n2);

    convw_kernel<7, 4><<<dim3(32, 64), 256, 0, stream>>>(h2, k3, h3, stats + 2 * 128);
    norm_kernel <<<(unsigned)(n3 / 256), 256, 0, stream>>>(h3, stats + 2 * 128,
                                                           1.f / 65536.f, n3);

    convw_kernel<4, 2><<<dim3(8, 64), 256, 0, stream>>>(h3, k4, h4, stats + 3 * 128);

    maxpool_kernel<<<(unsigned)(64L * 64 * 64 / 256), 256, 0, stream>>>(
        h4, stats + 3 * 128, feat);

    readout_kernel<<<64, 256, 0, stream>>>(feat, w_ro, out);
}
